// CR8_reg_3stage_13975823582046
// MI455X (gfx1250) — compile-verified, matrix operand order checked by reading, not run on hardware
//
#include <hip/hip_runtime.h>
#include <hip/hip_bf16.h>

// MI455X / gfx1250: f16 WMMA GEMMs (v_wmma_f32_16x16x32_f16) for all 1x1 convs,
// class-binned WMMA GEMMs for CondMul stages, scalar tails.
// GEMMs templated on <K, CO, OUT16, LRELU> for branch-free unrolled epilogues.

typedef __attribute__((ext_vector_type(16))) _Float16 v16h;
typedef __attribute__((ext_vector_type(8)))  _Float16 h8;
typedef __attribute__((ext_vector_type(8)))  float    v8f;

#define NPIX   35840            // 160*224
#define NTILES (NPIX / 16)      // 2240

static __device__ __forceinline__ v16h pack16(h8 lo, h8 hi) {
    v16h r;
#pragma unroll
    for (int i = 0; i < 8; ++i) { r[i] = lo[i]; r[i + 8] = hi[i]; }
    return r;
}

static __device__ __forceinline__ float lrelu_f(float x) {
    return x > 0.f ? x : 0.01f * x;
}

// ---------------- conversion / layout kernels ----------------

__global__ void k_convert(const float* __restrict__ src, _Float16* __restrict__ dst, int n) {
    int i = blockIdx.x * blockDim.x + threadIdx.x;
    int stride = gridDim.x * blockDim.x;
    for (; i < n; i += stride) dst[i] = (_Float16)src[i];
}

// src: [cls][Ci][Co] f32  ->  dst: [cls][Co][Ci] f16
__global__ void k_convert_tr(const float* __restrict__ src, _Float16* __restrict__ dst,
                             int cls, int Ci, int Co) {
    int n = cls * Ci * Co;
    int i = blockIdx.x * blockDim.x + threadIdx.x;
    int stride = gridDim.x * blockDim.x;
    for (; i < n; i += stride) {
        int c = i / (Ci * Co);
        int r = i - c * (Ci * Co);
        int ci = r / Co;
        int co = r - ci * Co;
        dst[((size_t)c * Co + co) * Ci + ci] = (_Float16)src[i];
    }
}

// x_in [128, NPIX] f32 (channel-major) -> xT [NPIX, 128] f16 (pixel-major)
__global__ void k_transpose_x(const float* __restrict__ x, _Float16* __restrict__ xT) {
    int idx = blockIdx.x * blockDim.x + threadIdx.x;   // exact grid: 128*NPIX threads
    int c = idx / NPIX;
    int p = idx - c * NPIX;
    xT[(size_t)p * 128 + c] = (_Float16)x[idx];
}

// ---------------- fragment loaders ----------------

template <int K, int NKT>
static __device__ __forceinline__ void load_frags(const _Float16* __restrict__ base,
                                                  v16h (&f)[NKT]) {
#pragma unroll
    for (int kt = 0; kt < NKT; ++kt) {
        h8 lo = *(const h8*)(base + kt * 32);
        h8 h2 = *(const h8*)(base + kt * 32 + 16);
        f[kt] = pack16(lo, h2);
    }
}

// ---------------- dense WMMA GEMM:  out[N,CO] = act(A[N,K] @ W[CO,K]^T + b) ----------------

template <int K, int CO, bool OUT16, bool LRELU>
__global__ __launch_bounds__(256) void k_gemm(const _Float16* __restrict__ A,
                                              const _Float16* __restrict__ W,
                                              const float* __restrict__ bias,
                                              _Float16* __restrict__ out16,
                                              float* __restrict__ out32) {
    constexpr int NKT = K / 32;
    int wave = (blockIdx.x * blockDim.x + threadIdx.x) >> 5;
    if (wave >= NTILES) return;                    // wave-uniform
    int lane = threadIdx.x & 31;
    int m  = lane & 15;
    int hi = lane >> 4;

    // A fragment: lane row = tile pixel m; halves 0..7 = K{0..7|8..15}, 8..15 = K{16..23|24..31}
    const _Float16* arow = A + (size_t)(wave * 16 + m) * K + (hi ? 8 : 0);
    v16h a[NKT];
    load_frags<K, NKT>(arow, a);

#pragma unroll
    for (int ct = 0; ct < CO / 16; ++ct) {
        v8f acc = {};
        const _Float16* wrow = W + (size_t)(ct * 16 + m) * K + (hi ? 8 : 0);
        v16h b[NKT];
        load_frags<K, NKT>(wrow, b);
#pragma unroll
        for (int kt = 0; kt < NKT; ++kt)
            acc = __builtin_amdgcn_wmma_f32_16x16x32_f16(false, a[kt], false, b[kt],
                                                         (short)0, acc, false, false);
        int co = ct * 16 + m;
        float bb = bias[co];
#pragma unroll
        for (int r = 0; r < 8; ++r) {
            int row = wave * 16 + hi * 8 + r;
            float v = acc[r] + bb;
            if (LRELU) v = lrelu_f(v);
            size_t off = (size_t)row * CO + co;
            if (OUT16) out16[off] = (_Float16)v;
            else       out32[off] = v;
        }
    }
}

// ---------------- class-binned WMMA GEMM (CondMul) ----------------
// W: [cls][CO][K] f16, bias: [cls][CO] f32, jobs: triples {cls, rowStart, rowEnd} into ord[]

template <int K, int CO, bool OUT16, bool LRELU>
__global__ __launch_bounds__(256) void k_cgemm(const _Float16* __restrict__ A,
                                               const _Float16* __restrict__ W,
                                               const float* __restrict__ bias,
                                               const int* __restrict__ ord,
                                               const int* __restrict__ jobs,
                                               const int* __restrict__ njobs,
                                               int maxJobs,
                                               _Float16* __restrict__ out16,
                                               float* __restrict__ out32) {
    constexpr int NKT = K / 32;
    int w = (blockIdx.x * blockDim.x + threadIdx.x) >> 5;
    if (w >= maxJobs) return;                      // wave-uniform
    if (w >= njobs[0]) return;                     // wave-uniform
    int lane = threadIdx.x & 31;
    int m  = lane & 15;
    int hi = lane >> 4;

    int cls  = jobs[3 * w];
    int rs   = jobs[3 * w + 1];
    int re   = jobs[3 * w + 2];
    int rows = re - rs;                            // 1..16
    int myr  = m < rows ? m : rows - 1;            // clamp (dup rows computed, not stored)
    int pix  = ord[rs + myr];

    const _Float16* arow = A + (size_t)pix * K + (hi ? 8 : 0);
    v16h a[NKT];
    load_frags<K, NKT>(arow, a);

    const _Float16* Wc = W + (size_t)cls * CO * K;
#pragma unroll
    for (int ct = 0; ct < CO / 16; ++ct) {
        v8f acc = {};
        const _Float16* wrow = Wc + (size_t)(ct * 16 + m) * K + (hi ? 8 : 0);
        v16h b[NKT];
        load_frags<K, NKT>(wrow, b);
#pragma unroll
        for (int kt = 0; kt < NKT; ++kt)
            acc = __builtin_amdgcn_wmma_f32_16x16x32_f16(false, a[kt], false, b[kt],
                                                         (short)0, acc, false, false);
        int co = ct * 16 + m;
        float bb = bias[(size_t)cls * CO + co];
#pragma unroll
        for (int r = 0; r < 8; ++r) {
            int row = hi * 8 + r;
            int pp = __shfl(pix, row);             // unconditional: all lanes active here
            if (row < rows) {
                float v = acc[r] + bb;
                if (LRELU) v = lrelu_f(v);
                size_t off = (size_t)pp * CO + co;
                if (OUT16) out16[off] = (_Float16)v;
                else       out32[off] = v;
            }
        }
    }
}

// ---------------- binning kernels ----------------

__global__ void k_count(const int* __restrict__ inds, int* __restrict__ cnt, int n) {
    int p = blockIdx.x * blockDim.x + threadIdx.x;
    if (p < n) atomicAdd(&cnt[inds[p]], 1);
}

__global__ void k_scan_jobs(const int* __restrict__ cnt, int* __restrict__ cursor,
                            int* __restrict__ jobs, int* __restrict__ njobs, int C) {
    __shared__ int sh[256];
    int t = threadIdx.x;
    int v = (t < C) ? cnt[t] : 0;
    sh[t] = v;
    __syncthreads();
    for (int off = 1; off < 256; off <<= 1) {
        int x = (t >= off) ? sh[t - off] : 0;
        __syncthreads();
        sh[t] += x;
        __syncthreads();
    }
    int start = sh[t] - v;                         // exclusive scan
    if (t < C) {
        cursor[t] = start;
        int nt = (v + 15) >> 4;
        for (int k = 0; k < nt; ++k) {
            int j = atomicAdd(njobs, 1);
            jobs[3 * j]     = t;
            jobs[3 * j + 1] = start + 16 * k;
            int e = start + 16 * (k + 1);
            jobs[3 * j + 2] = e < start + v ? e : start + v;
        }
    }
}

__global__ void k_scatter(const int* __restrict__ inds, int* __restrict__ cursor,
                          int* __restrict__ ord, int n) {
    int p = blockIdx.x * blockDim.x + threadIdx.x;
    if (p < n) {
        int pos = atomicAdd(&cursor[inds[p]], 1);
        ord[pos] = p;
    }
}

// ---------------- argmax / tail kernels ----------------

__global__ void k_argmax1(const float* __restrict__ t, int* __restrict__ inds1, int n) {
    int p = blockIdx.x * blockDim.x + threadIdx.x;
    if (p >= n) return;
    const float* row = t + (size_t)p * 16;
    float best = row[0]; int bi = 0;
    for (int i = 1; i < 16; ++i) { float v = row[i]; if (v > best) { best = v; bi = i; } }
    inds1[p] = bi;
}

__global__ void k_argmax2(const float* __restrict__ t, const int* __restrict__ inds1,
                          int* __restrict__ inds12, int n) {
    int p = blockIdx.x * blockDim.x + threadIdx.x;
    if (p >= n) return;
    const float* row = t + (size_t)p * 32;
    float best = row[0]; int bi = 0;
    for (int i = 1; i < 32; ++i) { float v = row[i]; if (v > best) { best = v; bi = i; } }
    int v12 = inds1[p] * 16 + (bi - 8);
    v12 = v12 < 0 ? 0 : (v12 > 255 ? 255 : v12);
    inds12[p] = v12;
}

__global__ void k_argmax3(const float* __restrict__ t, const int* __restrict__ inds12,
                          int* __restrict__ inds123, int* __restrict__ indsSup, int n) {
    int p = blockIdx.x * blockDim.x + threadIdx.x;
    if (p >= n) return;
    const float* row = t + (size_t)p * 32;
    float best = row[0]; int bi = 0;
    for (int i = 1; i < 32; ++i) { float v = row[i]; if (v > best) { best = v; bi = i; } }
    int v = inds12[p] * 16 + (bi - 8);
    v = v < 0 ? 0 : (v > 4095 ? 4095 : v);
    inds123[p] = v;
    indsSup[p] = v / 512;
}

__global__ void k_mask(const _Float16* __restrict__ m16, const float* __restrict__ w,
                       const float* __restrict__ b, float* __restrict__ out, int n) {
    int p = blockIdx.x * blockDim.x + threadIdx.x;
    if (p >= n) return;
    float acc = b[0];
    const _Float16* row = m16 + (size_t)p * 16;
    for (int i = 0; i < 16; ++i) acc += (float)row[i] * w[i];
    out[p] = lrelu_f(acc);
}

__global__ void k_reg3(const _Float16* __restrict__ t, const int* __restrict__ inds123,
                       const float* __restrict__ r3W, const float* __restrict__ r3b,
                       float* __restrict__ out, int n) {
    int p = blockIdx.x * blockDim.x + threadIdx.x;
    if (p >= n) return;
    int cls = inds123[p];
    float acc = r3b[cls];
    const _Float16* row = t + (size_t)p * 32;
    const float* wr = r3W + (size_t)cls * 32;
    for (int i = 0; i < 32; ++i) acc += (float)row[i] * wr[i];
    out[p] = ((float)cls + acc) * (1.0f / 4096.0f);
}

// ---------------- host ----------------

extern "C" void kernel_launch(void* const* d_in, const int* in_sizes, int n_in,
                              void* d_out, int out_size, void* d_ws, size_t ws_size,
                              hipStream_t stream) {
    (void)in_sizes; (void)n_in; (void)out_size; (void)ws_size;
    char* ws = (char*)d_ws;
    size_t o = 0;
    auto alloc = [&](size_t bytes) -> char* {
        char* r = ws + o;
        o = (o + bytes + 255) & ~(size_t)255;
        return r;
    };

    _Float16* xT   = (_Float16*)alloc((size_t)NPIX * 128 * 2);
    _Float16* buf1 = (_Float16*)alloc((size_t)NPIX * 128 * 2);  // feat
    _Float16* buf2 = (_Float16*)alloc((size_t)NPIX * 128 * 2);  // tmp / xr
    _Float16* tA   = (_Float16*)alloc((size_t)NPIX * 32 * 2);
    _Float16* tB   = (_Float16*)alloc((size_t)NPIX * 32 * 2);
    float*    tF   = (float*)   alloc((size_t)NPIX * 32 * 4);
    int* inds1   = (int*)alloc((size_t)NPIX * 4);
    int* inds12  = (int*)alloc((size_t)NPIX * 4);
    int* inds123 = (int*)alloc((size_t)NPIX * 4);
    int* indsSup = (int*)alloc((size_t)NPIX * 4);
    int* ord     = (int*)alloc((size_t)NPIX * 4);
    int* cnt     = (int*)alloc(256 * 4);
    int* cursor  = (int*)alloc(256 * 4);
    int* jobs    = (int*)alloc((size_t)(NTILES + 256) * 3 * 4);
    int* njobs   = (int*)alloc(4);

    _Float16* w_bb1  = (_Float16*)alloc(16384 * 2);
    _Float16* w_bb2  = (_Float16*)alloc(16384 * 2);
    _Float16* w_bb3  = (_Float16*)alloc(16384 * 2);
    _Float16* w_msk1 = (_Float16*)alloc(4096 * 2);
    _Float16* w_msk2 = (_Float16*)alloc(512 * 2);
    _Float16* w_c10  = (_Float16*)alloc(4096 * 2);
    _Float16* w_c20  = (_Float16*)alloc(1024 * 2);
    _Float16* w_c30  = (_Float16*)alloc(512 * 2);
    _Float16* w_r1   = (_Float16*)alloc(16384 * 2);
    _Float16* w_c11  = (_Float16*)alloc((size_t)16 * 128 * 32 * 2);
    _Float16* w_c21  = (_Float16*)alloc((size_t)16 * 32 * 32 * 2);
    _Float16* w_c31  = (_Float16*)alloc((size_t)16 * 32 * 32 * 2);
    _Float16* w_c12  = (_Float16*)alloc((size_t)256 * 128 * 32 * 2);
    _Float16* w_c22  = (_Float16*)alloc((size_t)256 * 32 * 32 * 2);
    _Float16* w_c32  = (_Float16*)alloc((size_t)256 * 32 * 32 * 2);
    _Float16* w_r2   = (_Float16*)alloc((size_t)8 * 128 * 32 * 2);

    auto fin = [&](int i) { return (const float*)d_in[i]; };
    auto cdiv = [](int a, int b) { return (a + b - 1) / b; };

    // --- weight conversion ---
    auto conv = [&](int i, _Float16* dst, int n) {
        k_convert<<<cdiv(n, 256), 256, 0, stream>>>(fin(i), dst, n);
    };
    conv(1, w_bb1, 16384);  conv(3, w_bb2, 16384);  conv(5, w_bb3, 16384);
    conv(7, w_msk1, 4096);  conv(9, w_msk2, 512);
    conv(13, w_c10, 4096);  conv(15, w_c20, 1024);  conv(17, w_c30, 512);
    conv(31, w_r1, 16384);
    k_convert_tr<<<cdiv(16 * 128 * 32, 256), 256, 0, stream>>>(fin(19), w_c11, 16, 128, 32);
    k_convert_tr<<<cdiv(16 * 32 * 32, 256), 256, 0, stream>>>(fin(21), w_c21, 16, 32, 32);
    k_convert_tr<<<cdiv(16 * 32 * 32, 256), 256, 0, stream>>>(fin(23), w_c31, 16, 32, 32);
    k_convert_tr<<<cdiv(256 * 128 * 32, 256), 256, 0, stream>>>(fin(25), w_c12, 256, 128, 32);
    k_convert_tr<<<cdiv(256 * 32 * 32, 256), 256, 0, stream>>>(fin(27), w_c22, 256, 32, 32);
    k_convert_tr<<<cdiv(256 * 32 * 32, 256), 256, 0, stream>>>(fin(29), w_c32, 256, 32, 32);
    k_convert_tr<<<cdiv(8 * 128 * 32, 256), 256, 0, stream>>>(fin(33), w_r2, 8, 128, 32);

    // --- input transpose to pixel-major f16 ---
    k_transpose_x<<<(128 * NPIX) / 256, 256, 0, stream>>>(fin(0), xT);

    const int GB = NTILES / 8;   // 280 blocks of 8 waves (dense gemms)
    const int GE = cdiv(NPIX, 256);

    // --- backbone: feat = lrelu(lrelu(lrelu(x@bb1)@bb2)@bb3) ---
    k_gemm<128, 128, true, true><<<GB, 256, 0, stream>>>(xT,   w_bb1, fin(2), buf1, nullptr);
    k_gemm<128, 128, true, true><<<GB, 256, 0, stream>>>(buf1, w_bb2, fin(4), buf2, nullptr);
    k_gemm<128, 128, true, true><<<GB, 256, 0, stream>>>(buf2, w_bb3, fin(6), buf1, nullptr); // feat=buf1

    // --- mask head ---
    k_gemm<128, 32, true, true><<<GB, 256, 0, stream>>>(xT, w_msk1, fin(8), tA, nullptr);
    k_gemm<32, 16, true, true><<<GB, 256, 0, stream>>>(tA, w_msk2, fin(10), tB, nullptr);
    k_mask<<<GE, 256, 0, stream>>>(tB, fin(11), fin(12), (float*)d_out + NPIX, NPIX);

    // --- stage 1 (dense classifier) ---
    k_gemm<128, 32, true, true><<<GB, 256, 0, stream>>>(buf1, w_c10, fin(14), tA, nullptr);
    k_gemm<32, 32, true, true><<<GB, 256, 0, stream>>>(tA, w_c20, fin(16), tB, nullptr);
    k_gemm<32, 16, false, false><<<GB, 256, 0, stream>>>(tB, w_c30, fin(18), nullptr, tF);
    k_argmax1<<<GE, 256, 0, stream>>>(tF, inds1, NPIX);

    auto bin = [&](const int* inds, int C) {
        hipMemsetAsync(cnt, 0, 256 * 4, stream);
        hipMemsetAsync(njobs, 0, 4, stream);
        k_count<<<GE, 256, 0, stream>>>(inds, cnt, NPIX);
        k_scan_jobs<<<1, 256, 0, stream>>>(cnt, cursor, jobs, njobs, C);
        k_scatter<<<GE, 256, 0, stream>>>(inds, cursor, ord, NPIX);
    };

    // --- stage 2: CondMul over 16 classes routed by inds1 ---
    bin(inds1, 16);
    {
        int mj = NTILES + 16, gc = cdiv(mj * 32, 256);
        k_cgemm<128, 32, true, true><<<gc, 256, 0, stream>>>(buf1, w_c11, fin(20), ord, jobs, njobs, mj, tA, nullptr);
        k_cgemm<32, 32, true, true><<<gc, 256, 0, stream>>>(tA, w_c21, fin(22), ord, jobs, njobs, mj, tB, nullptr);
        k_cgemm<32, 32, false, false><<<gc, 256, 0, stream>>>(tB, w_c31, fin(24), ord, jobs, njobs, mj, nullptr, tF);
    }
    k_argmax2<<<GE, 256, 0, stream>>>(tF, inds1, inds12, NPIX);

    // --- stage 3: CondMul over 256 classes routed by inds12 ---
    bin(inds12, 256);
    {
        int mj = NTILES + 256, gc = cdiv(mj * 32, 256);
        k_cgemm<128, 32, true, true><<<gc, 256, 0, stream>>>(buf1, w_c12, fin(26), ord, jobs, njobs, mj, tA, nullptr);
        k_cgemm<32, 32, true, true><<<gc, 256, 0, stream>>>(tA, w_c22, fin(28), ord, jobs, njobs, mj, tB, nullptr);
        k_cgemm<32, 32, false, false><<<gc, 256, 0, stream>>>(tB, w_c32, fin(30), ord, jobs, njobs, mj, nullptr, tF);
    }
    k_argmax3<<<GE, 256, 0, stream>>>(tF, inds12, inds123, indsSup, NPIX);

    // --- regression head ---
    k_gemm<128, 128, true, true><<<GB, 256, 0, stream>>>(xT, w_r1, fin(32), buf2, nullptr); // xr
    bin(indsSup, 8);
    {
        int mj = NTILES + 8, gc = cdiv(mj * 32, 256);
        k_cgemm<128, 32, true, true><<<gc, 256, 0, stream>>>(buf2, w_r2, fin(34), ord, jobs, njobs, mj, tA, nullptr);
    }
    k_reg3<<<GE, 256, 0, stream>>>(tA, inds123, fin(35), fin(36), (float*)d_out, NPIX);
}
